// G_Synthesis_Skip_21251498181275
// MI455X (gfx1250) — matrix operand order repaired and checked by reading, not compiled, not run on hardware
//
#include <hip/hip_runtime.h>
#include <hip/hip_bf16.h>
#include <math.h>

typedef __attribute__((ext_vector_type(16))) _Float16 v16h;
typedef __attribute__((ext_vector_type(8)))  _Float16 v8h;
typedef __attribute__((ext_vector_type(8)))  float    v8f;

static constexpr int   BN     = 16;      // batch
static constexpr int   SDIM   = 512;     // style dim
static constexpr float LSLOPE = 0.2f;
static constexpr float ASCALE = 1.41421356237309515f; // sqrt(2)

// =====================================================================
// Elementwise / prep kernels (bandwidth-trivial compared to the convs)
// =====================================================================

__global__ void k_nchw_to_nhwc(const float* __restrict__ x, _Float16* __restrict__ y,
                               int C, int H, size_t n) {
  size_t i = (size_t)blockIdx.x * blockDim.x + threadIdx.x;
  if (i >= n) return;
  int w = (int)(i % H); size_t r = i / H;
  int h = (int)(r % H); r /= H;
  int c = (int)(r % C); int b = (int)(r / C);
  y[(((size_t)b * H + h) * H + w) * C + c] = (_Float16)x[i];
}

// s[b,ci] = style[b,:] @ mod_w[:,ci] / sqrt(512) + mod_b[ci]
__global__ void k_style_affine(const float* __restrict__ st,   // [B,512]
                               const float* __restrict__ mw,   // [512,Cin]
                               const float* __restrict__ mb,   // [Cin]
                               float* __restrict__ s, int Cin, size_t n) {
  size_t i = (size_t)blockIdx.x * blockDim.x + threadIdx.x;
  if (i >= n) return;
  int ci = (int)(i % Cin); int b = (int)(i / Cin);
  float acc = 0.f;
  for (int k = 0; k < SDIM; ++k)
    acc += st[(size_t)b * SDIM + k] * mw[(size_t)k * Cin + ci];
  s[i] = acc * 0.04419417382415922f + mb[ci];   // 1/sqrt(512)
}

// Weight panel: bmat[co][t*Cin+ci] (f16, eq-LR scaled; up-mode applies the
// flip+transpose that turns ConvTranspose2d into a dilated conv), plus
// wsum[co][ci] = sum_t w^2 for demodulation.
__global__ void k_prep_w(const float* __restrict__ w, _Float16* __restrict__ bmat,
                         float* __restrict__ wsum, int Cin, int Cout, int upmode,
                         size_t n) {
  size_t i = (size_t)blockIdx.x * blockDim.x + threadIdx.x;
  if (i >= n) return;                              // n = Cout*Cin
  int ci = (int)(i % Cin); int co = (int)(i / Cin);
  float eq = rsqrtf((float)(Cin * 9));
  float ss = 0.f;
  const size_t Ktot = (size_t)9 * Cin;
  for (int t = 0; t < 9; ++t) {
    float v;
    if (!upmode) {
      v = w[((size_t)co * Cin + ci) * 9 + t] * eq;           // [Cout,Cin,3,3]
    } else {
      int ky = t / 3, kx = t - ky * 3;
      int tf = (2 - ky) * 3 + (2 - kx);                       // spatial flip
      v = w[((size_t)ci * Cout + co) * 9 + tf] * eq;          // [Cin,Cout,3,3]
    }
    bmat[(size_t)co * Ktot + (size_t)t * Cin + ci] = (_Float16)v;
    ss += v * v;
  }
  wsum[i] = ss;
}

__global__ void k_demod(const float* __restrict__ wsum, const float* __restrict__ s,
                        float* __restrict__ d, int Cin, int Cout, size_t n) {
  size_t i = (size_t)blockIdx.x * blockDim.x + threadIdx.x;
  if (i >= n) return;                              // n = B*Cout
  int co = (int)(i % Cout); int b = (int)(i / Cout);
  float acc = 0.f;
  for (int ci = 0; ci < Cin; ++ci) {
    float sv = s[(size_t)b * Cin + ci];
    acc += wsum[(size_t)co * Cin + ci] * sv * sv;
  }
  d[i] = rsqrtf(acc + 1e-8f);
}

// xm = xh * s[b,c]   (style modulation folded into activations so the WMMA
// kernel can do pure 128-bit fragment loads)
__global__ void k_modulate(const _Float16* __restrict__ x, const float* __restrict__ s,
                           _Float16* __restrict__ y, int HW, int C, size_t n) {
  size_t i = (size_t)blockIdx.x * blockDim.x + threadIdx.x;
  if (i >= n) return;
  int c = (int)(i % C);
  int b = (int)(i / ((size_t)HW * C));
  y[i] = (_Float16)((float)x[i] * s[(size_t)b * C + c]);
}

// =====================================================================
// Core: modulated conv as implicit GEMM with V_WMMA_F32_16X16X32_F16.
//   M = output pixels (tile 16), N = Cout (tile 16), K = 9*Cin (step 32).
//   up==0: 3x3 SAME conv.  up==1: ConvTranspose2d(stride2) via dilated conv
//   (pad 2, zero-stuffed input), OH = 2*IH+1.
//
// The full B panel for this block's N-tile (16 x 9*Cin f16, <=147KB) is
// staged once into LDS and shared by all 4 waves; inner-loop B reads are
// ds_load_b128. The K loop is taps(9, fully unrolled) x chunks(Cin/32) so
// no runtime division remains in the hot path.
//
// A lane layout (16-bit A, 16x32): lanes<16 hold K={c..c+7, c+16..c+23};
// lanes>=16 the same +8 -> two contiguous 8-channel runs -> 2x b128 loads.
// B lane layout (32x16): same K order as A (lanes<16 K=0..7,16..23; lanes>=16 +8) -> 2x b128.
// =====================================================================
__global__ void k_modconv_wmma(const _Float16* __restrict__ xm,   // [B,IH,IW,Cin] modulated
                               const _Float16* __restrict__ bmat, // [Cout, 9*Cin]
                               const float* __restrict__ dmod,    // [B,Cout] (demod)
                               const float* __restrict__ actb,    // [Cout]
                               _Float16* __restrict__ y,          // [B,OH,OW,Cout]
                               int Cin, int Cout, int IH, int OH,
                               int up, int act) {
  const int IW = IH, OW = OH;
  const int Ktot = 9 * Cin;

  // ---- cooperative LDS stage of the 16 x Ktot B panel (all 128 threads) ----
  extern __shared__ __align__(16) _Float16 lsB[];
  {
    const _Float16* gB = bmat + (size_t)(blockIdx.y * 16) * Ktot;
    for (int r16 = 0; r16 < 16; ++r16) {
      for (int cc = threadIdx.x * 8; cc < Ktot; cc += 128 * 8) {
        *(v8h*)(lsB + r16 * Ktot + cc) =
            *(const v8h*)(gB + (size_t)r16 * Ktot + cc);
      }
    }
  }
  __syncthreads();

  const int lane = threadIdx.x & 31;
  const int wave = threadIdx.x >> 5;
  const int OHW  = OH * OW;
  const int numM = (OHW + 15) >> 4;
  const int tm   = blockIdx.x * 4 + wave;
  if (tm >= numM) return;                      // wave-uniform -> EXEC stays full
  const int half = lane >> 4;
  const int l15  = lane & 15;
  const int b    = blockIdx.z;
  const int co   = blockIdx.y * 16 + l15;      // B-fragment column
  int m = tm * 16 + l15;                       // A-fragment row (pixel)
  if (m >= OHW) m = OHW - 1;                   // clamp loads; stores are guarded
  const int oy = m / OW, ox = m - (m / OW) * OW;

  const _Float16* lrow = lsB + l15 * Ktot;     // this lane's B column
  const int aoff = half ? 8 : 0;               // A run channel offset
  const int boff = half ? 8 : 0;               // B run K offset

  v8f acc = {};
#pragma unroll
  for (int t = 0; t < 9; ++t) {
    const int ky = t / 3, kx = t - (t / 3) * 3;          // compile-time (unrolled)
    // per-tap A base pointer + validity (hoisted out of the channel loop)
    bool valid = false;
    const _Float16* arow = xm;
    if (!up) {
      int iy = oy + ky - 1, ix = ox + kx - 1;
      if ((unsigned)iy < (unsigned)IH && (unsigned)ix < (unsigned)IW) {
        valid = true;
        arow = xm + (((size_t)b * IH + iy) * IW + ix) * Cin;
      }
    } else {
      int dy = oy + ky - 2, dx = ox + kx - 2;            // dilated coords, pad 2
      if (((dy | dx) >= 0) && !((dy | dx) & 1)) {        // both >=0 and even
        int iy = dy >> 1, ix = dx >> 1;
        if (iy < IH && ix < IW) {
          valid = true;
          arow = xm + (((size_t)b * IH + iy) * IW + ix) * Cin;
        }
      }
    }
    const int kbase = t * Cin;
#pragma unroll 2
    for (int c0 = 0; c0 < Cin; c0 += 32) {
      union { v16h v; v8h h[2]; } A, Bf;
      v8h z = {};
      A.h[0] = z; A.h[1] = z;
      if (valid) {
        A.h[0] = *(const v8h*)(arow + c0 + aoff);        // K = c .. c+7
        A.h[1] = *(const v8h*)(arow + c0 + aoff + 16);   // K = c+16 .. c+23
      }
      const int cb = kbase + c0 + boff;
      Bf.h[0] = *(const v8h*)(lrow + cb);                // ds_load_b128
      Bf.h[1] = *(const v8h*)(lrow + cb + 16);
      acc = __builtin_amdgcn_wmma_f32_16x16x32_f16(
          /*neg_a=*/false, A.v, /*neg_b=*/false, Bf.v,
          /*c_mod=*/(short)0, acc, /*reuse_a=*/false, /*reuse_b=*/false);
    }
  }

  // Epilogue: demod, (bias + leaky-relu * sqrt2), store f16 NHWC.
  const float dm  = dmod[(size_t)b * Cout + co];
  const float bia = act ? actb[co] : 0.f;
#pragma unroll
  for (int j = 0; j < 8; ++j) {
    int mo = tm * 16 + j + (half ? 8 : 0);     // D layout: VGPR j -> row j(+8)
    if (mo < OHW) {
      float v = acc[j] * dm;
      if (act) { v += bia; v = (v > 0.f ? v : LSLOPE * v) * ASCALE; }
      int yy = mo / OW, xx = mo - yy * OW;
      y[(((size_t)b * OH + yy) * OW + xx) * Cout + co] = (_Float16)v;
    }
  }
}

// =====================================================================
// FIR [1,3,3,1]x[1,3,3,1]/16 blur (up=1, pad(1,1)) fused with bias+lrelu.
// Input (2H+1)^2, output (2H)^2.
// =====================================================================
__global__ void k_blur1_act(const _Float16* __restrict__ x, const float* __restrict__ actb,
                            _Float16* __restrict__ y, int C, int IHs, int OHs, size_t n) {
  size_t i = (size_t)blockIdx.x * blockDim.x + threadIdx.x;
  if (i >= n) return;
  int c = (int)(i % C); size_t r = i / C;
  int ox = (int)(r % OHs); r /= OHs;
  int oy = (int)(r % OHs); int b = (int)(r / OHs);
  const float a4[4] = {1.f, 3.f, 3.f, 1.f};
  float s = 0.f;
  for (int ky = 0; ky < 4; ++ky) {
    int iy = oy + ky - 1;
    if ((unsigned)iy >= (unsigned)IHs) continue;
    for (int kx = 0; kx < 4; ++kx) {
      int ix = ox + kx - 1;
      if ((unsigned)ix >= (unsigned)IHs) continue;
      s += a4[ky] * a4[kx] *
           (float)x[(((size_t)b * IHs + iy) * IHs + ix) * C + c];
    }
  }
  s = s * (1.f / 16.f) + actb[c];
  s = (s > 0.f ? s : LSLOPE * s) * ASCALE;
  y[i] = (_Float16)s;
}

// FIR 2x upsample of the RGB skip (up=2, pad(2,1)), NCHW f32. out = 2H.
__global__ void k_fir_up2_rgb(const float* __restrict__ x, float* __restrict__ y,
                              int H, int OHs, size_t n) {
  size_t i = (size_t)blockIdx.x * blockDim.x + threadIdx.x;
  if (i >= n) return;
  int ox = (int)(i % OHs); size_t r = i / OHs;
  int oy = (int)(r % OHs); r /= OHs;
  int c  = (int)(r % 3);   int b = (int)(r / 3);
  const float a4[4] = {1.f, 3.f, 3.f, 1.f};
  float s = 0.f;
  for (int ky = 0; ky < 4; ++ky) {
    int dy = oy + ky - 2;
    if (dy < 0 || (dy & 1)) continue;
    int iy = dy >> 1; if (iy >= H) continue;
    for (int kx = 0; kx < 4; ++kx) {
      int dx = ox + kx - 2;
      if (dx < 0 || (dx & 1)) continue;
      int ix = dx >> 1; if (ix >= H) continue;
      s += a4[ky] * a4[kx] * x[(((size_t)b * 3 + c) * H + iy) * H + ix];
    }
  }
  y[i] = s * (1.f / 16.f);   // FIR_GAIN=4 folded: (a/8)(a/8)*4 = a*a/16
}

// to_rgb: 1x1 modulated conv (no demod), 3 outputs -> VALU dot (N=3 is far
// too thin for a 16-wide WMMA tile). Optionally accumulates onto skip.
__global__ void k_to_out(const _Float16* __restrict__ xh, const float* __restrict__ s,
                         const float* __restrict__ w, const float* __restrict__ bias,
                         float* __restrict__ rgb, int Cin, int H, int addPrev,
                         size_t n) {
  size_t i = (size_t)blockIdx.x * blockDim.x + threadIdx.x;
  if (i >= n) return;
  int ox = (int)(i % H); size_t r = i / H;
  int oy = (int)(r % H); r /= H;
  int c  = (int)(r % 3); int b = (int)(r / 3);
  const float eq = rsqrtf((float)Cin);
  const _Float16* px = xh + (((size_t)b * H + oy) * H + ox) * Cin;
  const float* sb = s + (size_t)b * Cin;
  const float* wc = w + (size_t)c * Cin;
  float acc = 0.f;
  for (int ci = 0; ci < Cin; ++ci) acc += (float)px[ci] * sb[ci] * wc[ci];
  acc = acc * eq + bias[c];
  if (addPrev) acc += rgb[i];
  rgb[i] = acc;
}

__global__ void k_copy_f32(const float* __restrict__ x, float* __restrict__ y, size_t n) {
  size_t i = (size_t)blockIdx.x * blockDim.x + threadIdx.x;
  if (i < n) y[i] = x[i];
}

// =====================================================================
// Host orchestration
// =====================================================================
static inline dim3 g1(size_t n) { return dim3((unsigned)((n + 255) / 256)); }

extern "C" void kernel_launch(void* const* d_in, const int* in_sizes, int n_in,
                              void* d_out, int out_size, void* d_ws, size_t ws_size,
                              hipStream_t stream) {
  (void)in_sizes; (void)n_in; (void)ws_size;
  const float* init_feat = (const float*)d_in[0];          // [16,512,4,4]
  const float* style_all = (const float*)d_in[1];          // [14,16,512]

  // params flattened in jax tree order (dict keys sorted):
  //  blocks[0..5]: btk{act_b,mod_b,mod_w,w}, to_out{bias,mod_b,mod_w,w},
  //                up{act_b,mod_b,mod_w,w};  first: conv{...}, to_out{...}
  struct CP { const float *act_b, *mod_b, *mod_w, *w; };
  struct OP { const float *bias,  *mod_b, *mod_w, *w; };
  int p = 2;
  auto nx = [&]() { return (const float*)d_in[p++]; };
  CP btk[6], up[6]; OP tout[6];
  for (int k = 0; k < 6; ++k) {
    btk[k]  = { nx(), nx(), nx(), nx() };
    tout[k] = { nx(), nx(), nx(), nx() };
    up[k]   = { nx(), nx(), nx(), nx() };
  }
  CP firstc = { nx(), nx(), nx(), nx() };
  OP firsto = { nx(), nx(), nx(), nx() };

  // ---- workspace carve ----
  char* wsb = (char*)d_ws; size_t off = 0;
  auto carve = [&](size_t bytes) -> void* {
    void* r = wsb + off; off += (bytes + 255) & ~(size_t)255; return r;
  };
  const size_t MAXACT = (size_t)BN * 257 * 257 * 32;        // largest NHWC tensor
  _Float16* act[3];
  for (int k = 0; k < 3; ++k) act[k] = (_Float16*)carve(MAXACT * sizeof(_Float16));
  _Float16* bmat = (_Float16*)carve((size_t)512 * 512 * 9 * sizeof(_Float16));
  float* wsum = (float*)carve((size_t)512 * 512 * sizeof(float));
  float* sA   = (float*)carve((size_t)BN * 512 * sizeof(float));
  float* dmod = (float*)carve((size_t)BN * 512 * sizeof(float));
  float* rgbA = (float*)carve((size_t)BN * 3 * 256 * 256 * sizeof(float));
  float* rgbB = (float*)carve((size_t)BN * 3 * 256 * 256 * sizeof(float));

  const int cinB[6]  = {512, 512, 512, 256, 128, 64};
  const int coutB[6] = {512, 512, 256, 128, 64, 32};
  const int ihB[6]   = {4, 8, 16, 32, 64, 128};

  auto shmemB = [](int Cin) { return (size_t)16 * 9 * Cin * sizeof(_Float16); };

  // ---- first conv (4x4, 512->512) ----
  {
    size_t n = (size_t)BN * 512 * 4 * 4;
    k_nchw_to_nhwc<<<g1(n), 256, 0, stream>>>(init_feat, act[0], 512, 4, n);
    k_style_affine<<<g1((size_t)BN * 512), 256, 0, stream>>>(
        style_all + 0, firstc.mod_w, firstc.mod_b, sA, 512, (size_t)BN * 512);
    k_prep_w<<<g1((size_t)512 * 512), 256, 0, stream>>>(
        firstc.w, bmat, wsum, 512, 512, 0, (size_t)512 * 512);
    k_demod<<<g1((size_t)BN * 512), 256, 0, stream>>>(wsum, sA, dmod, 512, 512,
                                                      (size_t)BN * 512);
    k_modulate<<<g1(n), 256, 0, stream>>>(act[0], sA, act[1], 16, 512, n);
    int numM = (16 + 15) / 16;
    dim3 g((numM + 3) / 4, 512 / 16, BN);
    k_modconv_wmma<<<g, 128, shmemB(512), stream>>>(act[1], bmat, dmod,
                                                    firstc.act_b, act[2],
                                                    512, 512, 4, 4, 0, 1);
  }
  int cur = 2;

  // first to_rgb (style[1])
  {
    k_style_affine<<<g1((size_t)BN * 512), 256, 0, stream>>>(
        style_all + (size_t)1 * BN * SDIM, firsto.mod_w, firsto.mod_b, sA, 512,
        (size_t)BN * 512);
    size_t n = (size_t)BN * 3 * 4 * 4;
    k_to_out<<<g1(n), 256, 0, stream>>>(act[cur], sA, firsto.w, firsto.bias,
                                        rgbA, 512, 4, 0, n);
  }
  float* rgbPrev = rgbA; float* rgbNext = rgbB;

  int si = 1;
  for (int k = 0; k < 6; ++k) {
    const int Ci = cinB[k], Co = coutB[k], IH = ihB[k];
    const int R = 2 * IH, OHu = 2 * IH + 1;
    const int a = (cur + 1) % 3, t = (cur + 2) % 3;

    // ---- upsample conv-transpose (WMMA, dilated), demod, no act ----
    k_style_affine<<<g1((size_t)BN * Ci), 256, 0, stream>>>(
        style_all + (size_t)si * BN * SDIM, up[k].mod_w, up[k].mod_b, sA, Ci,
        (size_t)BN * Ci);
    k_prep_w<<<g1((size_t)Co * Ci), 256, 0, stream>>>(up[k].w, bmat, wsum, Ci, Co,
                                                      1, (size_t)Co * Ci);
    k_demod<<<g1((size_t)BN * Co), 256, 0, stream>>>(wsum, sA, dmod, Ci, Co,
                                                     (size_t)BN * Co);
    size_t nin = (size_t)BN * IH * IH * Ci;
    k_modulate<<<g1(nin), 256, 0, stream>>>(act[cur], sA, act[a], IH * IH, Ci, nin);
    {
      int numM = (OHu * OHu + 15) / 16;
      dim3 g((numM + 3) / 4, Co / 16, BN);
      k_modconv_wmma<<<g, 128, shmemB(Ci), stream>>>(act[a], bmat, dmod, nullptr,
                                                     act[t], Ci, Co, IH, OHu, 1, 0);
    }
    // FIR blur + bias + lrelu -> feat at R x R
    size_t nb = (size_t)BN * R * R * Co;
    k_blur1_act<<<g1(nb), 256, 0, stream>>>(act[t], up[k].act_b, act[cur], Co,
                                            OHu, R, nb);

    // ---- bottleneck 3x3 conv (WMMA), demod + act ----
    k_style_affine<<<g1((size_t)BN * Co), 256, 0, stream>>>(
        style_all + (size_t)(si + 1) * BN * SDIM, btk[k].mod_w, btk[k].mod_b, sA,
        Co, (size_t)BN * Co);
    k_prep_w<<<g1((size_t)Co * Co), 256, 0, stream>>>(btk[k].w, bmat, wsum, Co, Co,
                                                      0, (size_t)Co * Co);
    k_demod<<<g1((size_t)BN * Co), 256, 0, stream>>>(wsum, sA, dmod, Co, Co,
                                                     (size_t)BN * Co);
    k_modulate<<<g1(nb), 256, 0, stream>>>(act[cur], sA, act[a], R * R, Co, nb);
    {
      int numM = (R * R + 15) / 16;
      dim3 g((numM + 3) / 4, Co / 16, BN);
      k_modconv_wmma<<<g, 128, shmemB(Co), stream>>>(act[a], bmat, dmod,
                                                     btk[k].act_b, act[t],
                                                     Co, Co, R, R, 0, 1);
    }
    cur = t;

    // ---- RGB skip: fir 2x upsample + to_rgb add ----
    size_t nr = (size_t)BN * 3 * R * R;
    k_fir_up2_rgb<<<g1(nr), 256, 0, stream>>>(rgbPrev, rgbNext, IH, R, nr);
    k_style_affine<<<g1((size_t)BN * Co), 256, 0, stream>>>(
        style_all + (size_t)(si + 2) * BN * SDIM, tout[k].mod_w, tout[k].mod_b, sA,
        Co, (size_t)BN * Co);
    k_to_out<<<g1(nr), 256, 0, stream>>>(act[cur], sA, tout[k].w, tout[k].bias,
                                         rgbNext, Co, R, 1, nr);
    { float* tmp = rgbPrev; rgbPrev = rgbNext; rgbNext = tmp; }
    si += 2;
  }

  k_copy_f32<<<g1((size_t)out_size), 256, 0, stream>>>(rgbPrev, (float*)d_out,
                                                       (size_t)out_size);
}
